// perFrameVideo_Text_v1_34316788695747
// MI455X (gfx1250) — matrix operand order repaired and checked by reading, not compiled, not run on hardware
//
#include <hip/hip_runtime.h>
#include <hip/hip_bf16.h>
#include <math.h>

#define C_DIM 256
#define NHEAD 8
#define HD    32
#define ST    30
#define SA    40
#define S_TOK 70
#define B_SZ  2
#define T_FR  8
#define BT_N  16
#define NEGV  (-1e30f)

typedef __attribute__((ext_vector_type(16))) _Float16 v16h;
typedef __attribute__((ext_vector_type(8)))  float    v8f;

__device__ __forceinline__ v8f wmma16(v16h a, v16h b, v8f c) {
  return __builtin_amdgcn_wmma_f32_16x16x32_f16(false, a, false, b, (short)0, c,
                                                false, false);
}

// A operand (16x32 f16 chunk) gathered from LDS row-major [16][stride] at kbase.
// lane<16: row=lane, k={0..15}; lane>=16: row=lane-16, k={16..31}
// -> two 16B-contiguous runs per lane => 2x ds_load_b128
__device__ __forceinline__ v16h loadA(const _Float16* base, int stride, int kbase, int lane) {
  int m = lane & 15, hi = lane >> 4;
  const _Float16* p = base + m * stride + kbase + hi * 16;
  v16h a;
#pragma unroll
  for (int i = 0; i < 16; ++i) {
    int k = i;
    a[i] = p[k];
  }
  return a;
}

// B operand from pre-swizzled f16 tile: lane's 16 values are contiguous (32B)
__device__ __forceinline__ v16h loadBsw(const _Float16* tile, int lane) {
  return *(const v16h*)(tile + lane * 16);
}

// C/D layout: c[r] = C[(lane>=16?8:0)+r][lane&15]
__device__ __forceinline__ void storeC_f16(_Float16* dst, int stride, int colbase,
                                           int lane, v8f c) {
  int col = lane & 15, hi = lane >> 4;
#pragma unroll
  for (int r = 0; r < 8; ++r)
    dst[(hi * 8 + r) * stride + colbase + col] = (_Float16)c[r];
}

// ---------------------------------------------------------------------------
// Kernel 1: sine positional embedding + memory / k_in rows + key mask
// ---------------------------------------------------------------------------
__global__ void prep_kernel(const float* __restrict__ text, const float* __restrict__ amr,
                            const int* __restrict__ tpm, const int* __restrict__ apm,
                            float* __restrict__ k_in, float* __restrict__ memv,
                            int* __restrict__ maskI) {
  int bs = blockIdx.x;
  int b = bs / S_TOK, s = bs - b * S_TOK;
  int c = threadIdx.x;
  float mem, kin;
  if (s < ST) {
    mem = text[(b * ST + s) * C_DIM + c];
    float xe = 0.f, xl = 0.f;
    for (int j = 0; j < ST; ++j) {
      float nm = (tpm[b * ST + j] != 0) ? 0.f : 1.f;
      if (j <= s) xe += nm;
      xl += nm;
    }
    float xn = xe / (xl + 1e-6f) * 6.283185307179586f;
    int p = c >> 1;
    float dt = __powf(10000.f, (2.f * (float)p) / 256.f);
    float v = xn / dt;
    float posv = (c & 1) ? __cosf(v) : __sinf(v);
    kin = mem + posv;
  } else {
    mem = amr[(b * SA + (s - ST)) * C_DIM + c];
    kin = mem;
  }
  memv[bs * C_DIM + c] = mem;
  k_in[bs * C_DIM + c] = kin;
  if (c == 0) maskI[bs] = (s < ST) ? tpm[b * ST + s] : apm[b * SA + (s - ST)];
}

// ---------------------------------------------------------------------------
// Kernel 2: K = k_in @ Wk^T + bk ; V = memory @ Wv^T + bv (tiny)
// ---------------------------------------------------------------------------
__global__ void kvproj_kernel(const float* __restrict__ k_in, const float* __restrict__ memv,
                              const float* __restrict__ Wqkv, const float* __restrict__ bqkv,
                              float* __restrict__ Kp, float* __restrict__ Vp) {
  int blk = blockIdx.x;
  int mat = blk / (B_SZ * S_TOK);
  int bs  = blk - mat * (B_SZ * S_TOK);
  int c = threadIdx.x;
  const float* src = (mat == 0) ? (k_in + bs * C_DIM) : (memv + bs * C_DIM);
  const float* W = Wqkv + (C_DIM + mat * C_DIM) * C_DIM + c * C_DIM;
  __shared__ float srow[C_DIM];
  srow[c] = src[c];
  __syncthreads();
  float acc = bqkv[C_DIM + mat * C_DIM + c];
  for (int j = 0; j < C_DIM; ++j) acc += srow[j] * W[j];
  float* dst = (mat == 0) ? Kp : Vp;
  dst[bs * C_DIM + c] = acc;
}

// ---------------------------------------------------------------------------
// Kernel 3: pre-swizzle Wq / Wo / K / V into per-lane f16 WMMA B-operand tiles
// tile = 32 lanes x 16 halves = 1KB.  grid = 432 tiles, block = 512 threads.
//   t 0..127   : WqSw  tile (h*2+nt)*8+cc
//   t 128..255 : WoSw  tile (w*2+nt)*8+cc
//   t 256..335 : KSw   tile (b*8+h)*5+kt        (keys zero-padded to 80)
//   t 336..431 : VSw   tile ((b*8+h)*3+kc)*2+nt (keys zero-padded to 96)
// ---------------------------------------------------------------------------
__global__ void swizzle_kernel(const float* __restrict__ Wqkv, const float* __restrict__ Wo,
                               const float* __restrict__ Kp, const float* __restrict__ Vp,
                               _Float16* __restrict__ WqSw, _Float16* __restrict__ WoSw,
                               _Float16* __restrict__ KSw,  _Float16* __restrict__ VSw) {
  int t = blockIdx.x;
  int lane = threadIdx.x >> 4;
  int i = threadIdx.x & 15;
  int n = lane & 15, hi = lane >> 4;
  float val;
  _Float16* dst;
  if (t < 256) {
    const float* W = (t < 128) ? Wqkv : Wo;
    int tt = t & 127;
    int cc = tt & 7, nt = (tt >> 3) & 1, h = tt >> 4;
    val = W[(h * 32 + nt * 16 + n) * C_DIM + cc * 32 + hi * 16 + i];
    dst = ((t < 128) ? WqSw : WoSw) + tt * 512;
  } else if (t < 336) {
    int u = t - 256;
    int kt = u % 5; int h = (u / 5) & 7; int b = u / 40;
    int key = kt * 16 + n;
    int dim = h * 32 + hi * 16 + i;
    val = (key < S_TOK) ? Kp[(b * S_TOK + key) * C_DIM + dim] : 0.f;
    dst = KSw + u * 512;
  } else {
    int u = t - 336;
    int nt = u & 1; int kc = (u >> 1) % 3; int h = (u / 6) & 7; int b = u / 48;
    int key = kc * 32 + hi * 16 + i;
    int dim = h * 32 + nt * 16 + n;
    val = (key < S_TOK) ? Vp[(b * S_TOK + key) * C_DIM + dim] : 0.f;
    dst = VSw + u * 512;
  }
  dst[lane * 16 + i] = (_Float16)val;
}

// ---------------------------------------------------------------------------
// Kernel 4: fused per-16-query-tile cross attention (all WMMA) + gating
// ---------------------------------------------------------------------------
__global__ __launch_bounds__(256)
void xattn_kernel(const float* __restrict__ f0, const float* __restrict__ f1,
                  const float* __restrict__ f2, const float* __restrict__ f3,
                  const float* __restrict__ p0, const float* __restrict__ p1,
                  const float* __restrict__ p2, const float* __restrict__ p3,
                  const _Float16* __restrict__ WqSw, const float* __restrict__ bqkv,
                  const _Float16* __restrict__ WoSw, const float* __restrict__ bo,
                  const _Float16* __restrict__ KSw,  const _Float16* __restrict__ VSw,
                  const int* __restrict__ maskI, float* __restrict__ out) {
  __shared__ _Float16 sA[16 * 256];                       // x+p tile, f16
  __shared__ _Float16 sQh[NHEAD][16 * 32];                // per-head Q, f16
  __shared__ __align__(16) char sPT[NHEAD * 16 * 96 * 2]; // ph f16 [h][16][96] / tg_t f32 [256][16]
  __shared__ _Float16 sHo[16 * 256];                      // concat head outputs, f16

  _Float16* ph  = (_Float16*)sPT;
  float*    tgt = (float*)sPT;

  const int tid  = threadIdx.x;
  const int wave = tid >> 5;
  const int lane = tid & 31;

  int wg = blockIdx.x;
  int bt = wg / 532;
  int t  = wg - bt * 532;
  int l, q0base;
  if (t < 400)      { l = 0; q0base = t; }
  else if (t < 500) { l = 1; q0base = t - 400; }
  else if (t < 525) { l = 2; q0base = t - 500; }
  else              { l = 3; q0base = t - 525; }
  const int HW_[4] = {6400, 1600, 400, 100};
  const int HW = HW_[l];
  const int q0 = q0base * 16;
  const float* feat = (l == 0) ? f0 : (l == 1) ? f1 : (l == 2) ? f2 : f3;
  const float* pos  = (l == 0) ? p0 : (l == 1) ? p1 : (l == 2) ? p2 : p3;
  const long obase  = (l == 0) ? 0L : (l == 1) ? 26214400L : (l == 2) ? 32768000L : 34406400L;
  const int b = bt / T_FR;

  const float* fb = feat + (long)bt * C_DIM * HW;
  const float* pb = pos  + (long)bt * C_DIM * HW;
  const bool full = (q0 + 16 <= HW);

  // ---- stage A = (x + p) tile [16][256] into LDS ----
  if (full) {
    for (int u = tid; u < 1024; u += 256) {      // (c, q4): 256 x 4 float4s
      int c = u >> 2, q4 = (u & 3) << 2;
      float4 f = *(const float4*)(fb + c * HW + q0 + q4);
      float4 p = *(const float4*)(pb + c * HW + q0 + q4);
      sA[(q4 + 0) * 256 + c] = (_Float16)(f.x + p.x);
      sA[(q4 + 1) * 256 + c] = (_Float16)(f.y + p.y);
      sA[(q4 + 2) * 256 + c] = (_Float16)(f.z + p.z);
      sA[(q4 + 3) * 256 + c] = (_Float16)(f.w + p.w);
    }
  } else {
    for (int idx = tid; idx < 16 * 256; idx += 256) {
      int q = idx & 15, c = idx >> 4;
      int qg = q0 + q;
      float v = 0.f;
      if (qg < HW) v = fb[c * HW + qg] + pb[c * HW + qg];
      sA[q * 256 + c] = (_Float16)v;
    }
  }
  __syncthreads();

  // ---- Q projection for head h = wave ----
  const int h = wave;
  v8f qa0 = {}; v8f qa1 = {};
#pragma unroll
  for (int cc = 0; cc < 8; ++cc) {
    v16h a  = loadA(sA, 256, cc * 32, lane);
    v16h b0 = loadBsw(WqSw + (h * 16 + cc) * 512, lane);
    v16h b1 = loadBsw(WqSw + (h * 16 + 8 + cc) * 512, lane);
    qa0 = wmma16(a, b0, qa0);
    qa1 = wmma16(a, b1, qa1);
  }
  {
    int col = lane & 15, hi = lane >> 4;
#pragma unroll
    for (int r = 0; r < 8; ++r) {
      sQh[h][(hi * 8 + r) * 32 + col]      = (_Float16)(qa0[r] + bqkv[h * 32 + col]);
      sQh[h][(hi * 8 + r) * 32 + 16 + col] = (_Float16)(qa1[r] + bqkv[h * 32 + 16 + col]);
    }
  }
  __syncthreads();

  // ---- scores: Q[16x32] x K^T[32x16] per key tile ----
  v16h aq = loadA(&sQh[h][0], 32, 0, lane);
  v8f sc[5];
#pragma unroll
  for (int kt = 0; kt < 5; ++kt) {
    v16h bk = loadBsw(KSw + ((b * 8 + h) * 5 + kt) * 512, lane);
    v8f z = {};
    sc[kt] = wmma16(aq, bk, z);
  }

  // ---- masked softmax over 80 (70 real) keys ----
  const float scale = 0.17677669529663687f;  // 1/sqrt(32)
  {
    int col = lane & 15, hi = lane >> 4;
#pragma unroll
    for (int r = 0; r < 8; ++r) {
      float v[5]; float mx = NEGV;
#pragma unroll
      for (int kt = 0; kt < 5; ++kt) {
        int key = kt * 16 + col;
        float s = sc[kt][r] * scale;
        if (key >= S_TOK || maskI[b * S_TOK + key] != 0) s = NEGV;
        v[kt] = s; mx = fmaxf(mx, s);
      }
#pragma unroll
      for (int off = 1; off < 16; off <<= 1)
        mx = fmaxf(mx, __shfl_xor(mx, off, 16));
      float sum = 0.f;
#pragma unroll
      for (int kt = 0; kt < 5; ++kt) { v[kt] = __expf(v[kt] - mx); sum += v[kt]; }
#pragma unroll
      for (int off = 1; off < 16; off <<= 1)
        sum += __shfl_xor(sum, off, 16);
      float inv = 1.0f / sum;
#pragma unroll
      for (int kt = 0; kt < 5; ++kt)
        ph[(h * 16 + hi * 8 + r) * 96 + kt * 16 + col] = (_Float16)(v[kt] * inv);
    }
#pragma unroll
    for (int j = 0; j < 8; ++j) {           // zero padded attn columns 80..95
      int idx = lane * 8 + j;
      ph[(h * 16 + (idx >> 4)) * 96 + 80 + (idx & 15)] = (_Float16)0.f;
    }
  }
  __syncthreads();

  // ---- out_h[16x32] = attn[16x96] x V[96x32] (V pre-zeroed beyond 70 keys) ----
  v8f o0 = {}; v8f o1 = {};
#pragma unroll
  for (int kc = 0; kc < 3; ++kc) {
    v16h ap  = loadA(ph + (h * 16) * 96, 96, kc * 32, lane);
    v16h bv0 = loadBsw(VSw + (((b * 8 + h) * 3 + kc) * 2 + 0) * 512, lane);
    v16h bv1 = loadBsw(VSw + (((b * 8 + h) * 3 + kc) * 2 + 1) * 512, lane);
    o0 = wmma16(ap, bv0, o0);
    o1 = wmma16(ap, bv1, o1);
  }
  storeC_f16(sHo, 256, h * 32 + 0,  lane, o0);
  storeC_f16(sHo, 256, h * 32 + 16, lane, o1);
  __syncthreads();   // also guarantees all ph reads done (tg_t overlays ph)

  // ---- out-proj: wave handles output dims [wave*32, wave*32+32) ----
  v8f t0 = {}; v8f t1 = {};
#pragma unroll
  for (int cc = 0; cc < 8; ++cc) {
    v16h a  = loadA(sHo, 256, cc * 32, lane);
    v16h b0 = loadBsw(WoSw + (wave * 16 + cc) * 512, lane);
    v16h b1 = loadBsw(WoSw + (wave * 16 + 8 + cc) * 512, lane);
    t0 = wmma16(a, b0, t0);
    t1 = wmma16(a, b1, t1);
  }
  {  // transposed tg_t[d][q]: 8 consecutive f32 per run -> mergeable ds stores
    int col = lane & 15, hi = lane >> 4;
    int d0 = wave * 32 + col;
    int d1 = wave * 32 + 16 + col;
    float b0v = bo[d0], b1v = bo[d1];
#pragma unroll
    for (int r = 0; r < 8; ++r) tgt[d0 * 16 + hi * 8 + r] = t0[r] + b0v;
#pragma unroll
    for (int r = 0; r < 8; ++r) tgt[d1 * 16 + hi * 8 + r] = t1[r] + b1v;
  }
  __syncthreads();

  // ---- gate with x and store ----
  float* ob = out + obase + (long)bt * C_DIM * HW;
  if (full) {
    for (int u = tid; u < 1024; u += 256) {
      int c = u >> 2, q4 = (u & 3) << 2;
      float4 f  = *(const float4*)(fb + c * HW + q0 + q4);
      float4 tv = *(const float4*)(tgt + c * 16 + q4);
      float4 g;
      g.x = f.x * tv.x; g.y = f.y * tv.y; g.z = f.z * tv.z; g.w = f.w * tv.w;
      *(float4*)(ob + c * HW + q0 + q4) = g;
    }
  } else {
    for (int idx = tid; idx < 16 * 256; idx += 256) {
      int q = idx & 15, c = idx >> 4;
      int qg = q0 + q;
      if (qg < HW) ob[c * HW + qg] = fb[c * HW + qg] * tgt[c * 16 + q];
    }
  }
}

extern "C" void kernel_launch(void* const* d_in, const int* in_sizes, int n_in,
                              void* d_out, int out_size, void* d_ws, size_t ws_size,
                              hipStream_t stream) {
  (void)in_sizes; (void)n_in; (void)out_size; (void)ws_size;
  const float* vf[4] = {(const float*)d_in[0], (const float*)d_in[2],
                        (const float*)d_in[4], (const float*)d_in[6]};
  const float* vp[4] = {(const float*)d_in[1], (const float*)d_in[3],
                        (const float*)d_in[5], (const float*)d_in[7]};
  const float* text = (const float*)d_in[8];
  const float* amr  = (const float*)d_in[9];
  const int*   tpm  = (const int*)d_in[10];
  const int*   apm  = (const int*)d_in[11];
  const float* Wqkv = (const float*)d_in[12];
  const float* bqkv = (const float*)d_in[13];
  const float* Wo   = (const float*)d_in[14];
  const float* bo   = (const float*)d_in[15];

  char* ws = (char*)d_ws;
  float*    k_in = (float*)(ws + 0);          // 143360 B
  float*    memv = (float*)(ws + 143360);     // 143360 B
  float*    Kp   = (float*)(ws + 286720);     // 143360 B
  float*    Vp   = (float*)(ws + 430080);     // 143360 B
  int*     maskI = (int*)(ws + 573440);       //   1024 B
  _Float16* WqSw = (_Float16*)(ws + 574464);  // 131072 B
  _Float16* WoSw = (_Float16*)(ws + 705536);  // 131072 B
  _Float16* KSw  = (_Float16*)(ws + 836608);  //  81920 B
  _Float16* VSw  = (_Float16*)(ws + 918528);  //  98304 B

  prep_kernel<<<B_SZ * S_TOK, C_DIM, 0, stream>>>(text, amr, tpm, apm, k_in, memv, maskI);
  kvproj_kernel<<<2 * B_SZ * S_TOK, C_DIM, 0, stream>>>(k_in, memv, Wqkv, bqkv, Kp, Vp);
  swizzle_kernel<<<432, 512, 0, stream>>>(Wqkv, Wo, Kp, Vp, WqSw, WoSw, KSw, VSw);
  xattn_kernel<<<532 * BT_N, 256, 0, stream>>>(
      vf[0], vf[1], vf[2], vf[3], vp[0], vp[1], vp[2], vp[3],
      WqSw, bqkv, WoSw, bo, KSw, VSw, maskI, (float*)d_out);
}
